// EmbedClass_20237885899048
// MI455X (gfx1250) — matrix operand order repaired and checked by reading, not compiled, not run on hardware
//
#include <hip/hip_runtime.h>
#include <hip/hip_bf16.h>

typedef __attribute__((ext_vector_type(16))) __bf16 v16bf;
typedef __attribute__((ext_vector_type(8)))  __bf16 v8bf;
typedef __attribute__((ext_vector_type(8)))  float  v8f;
typedef __attribute__((ext_vector_type(4)))  float  v4f;

#define DIM 128
#define LDS_STRIDE 136   // 128 + 8 bf16 pad -> 272B row stride, kills bank conflicts

__global__ __launch_bounds__(256) void bilinear_diag_kernel(
    const float* __restrict__ S,   // (B, D) source
    const float* __restrict__ T,   // (B, D) target
    const float* __restrict__ Wt,  // (B,)   weight
    const float* __restrict__ Bs,  // (B,)   bias
    const float* __restrict__ Wm,  // (D, D) matrixOperator, row-major [k][n]
    float* __restrict__ partial,   // per-block partial sums of resid^2
    int numTiles)
{
    __shared__ __attribute__((aligned(32))) __bf16 WhT[DIM * LDS_STRIDE];
    __shared__ __attribute__((aligned(32))) __bf16 WlT[DIM * LDS_STRIDE];
    __shared__ float wavesum[8];

    // ---- one-time per workgroup: W -> transposed bf16 hi/lo in LDS ----
    for (int idx = threadIdx.x; idx < DIM * DIM; idx += blockDim.x) {
        int k = idx >> 7;          // row of W    (K dim of GEMM)
        int n = idx & (DIM - 1);   // col of W    (N dim of GEMM)
        float w = Wm[idx];
        __bf16 hi = (__bf16)w;
        __bf16 lo = (__bf16)(w - (float)hi);
        WhT[n * LDS_STRIDE + k] = hi;
        WlT[n * LDS_STRIDE + k] = lo;
    }
    __syncthreads();

    const int lane      = threadIdx.x & 31;
    const int waveInBlk = threadIdx.x >> 5;
    const int wave      = blockIdx.x * (blockDim.x >> 5) + waveInBlk;
    const int nwaves    = gridDim.x * (blockDim.x >> 5);
    const int h = lane >> 4;    // lane half: 0 = lanes 0-15, 1 = lanes 16-31
    const int c = lane & 15;    // row index (A) / column index (B,C) within tile

    float lsum = 0.0f;

    for (int tile = wave; tile < numTiles; tile += nwaves) {
        const float* srow = S + (size_t)(tile * 16 + c) * DIM;   // A row for this lane

        // prefetch next tile's S/T rows (global_prefetch_b8)
        int nt = tile + nwaves;
        if (nt < numTiles) {
            __builtin_prefetch(S + (size_t)(nt * 16 + c) * DIM, 0, 1);
            __builtin_prefetch(T + (size_t)(nt * 16 + c) * DIM, 0, 1);
        }

        // ---- A fragments for ALL 4 K-blocks, hoisted (16-bit A 16x32 layout) ----
        // lane half h: run0 = K[kb*32 + 8h .. +7], run1 = K[kb*32 + 16 + 8h .. +7]
        v16bf Ah[4], Al[4];
        #pragma unroll
        for (int kb = 0; kb < 4; ++kb) {
            const int k0 = kb * 32 + h * 8;
            const int k1 = k0 + 16;
            v4f f0a = *(const v4f*)(srow + k0);
            v4f f0b = *(const v4f*)(srow + k0 + 4);
            v4f f1a = *(const v4f*)(srow + k1);
            v4f f1b = *(const v4f*)(srow + k1 + 4);
            float fv[16];
            #pragma unroll
            for (int i = 0; i < 4; ++i) {
                fv[i]      = f0a[i];
                fv[4 + i]  = f0b[i];
                fv[8 + i]  = f1a[i];
                fv[12 + i] = f1b[i];
            }
            #pragma unroll
            for (int i = 0; i < 16; ++i) {
                __bf16 hi = (__bf16)fv[i];
                Ah[kb][i] = hi;
                Al[kb][i] = (__bf16)(fv[i] - (float)hi);
            }
        }

        // ---- N-block outer loop: ONE live accumulator, folded into rp[] ----
        float rp[8];
        #pragma unroll
        for (int r = 0; r < 8; ++r) rp[r] = 0.0f;
        const float* trow = T + (size_t)tile * 16 * DIM;

        #pragma unroll 2
        for (int j = 0; j < 8; ++j) {
            v8f acc = {};
            #pragma unroll
            for (int kb = 0; kb < 4; ++kb) {
                const int kB = kb * 32 + h * 8;
                const __bf16* bh = &WhT[(j * 16 + c) * LDS_STRIDE + kB];
                const __bf16* bl = &WlT[(j * 16 + c) * LDS_STRIDE + kB];
                v8bf bh0 = *(const v8bf*)bh;
                v8bf bh1 = *(const v8bf*)(bh + 16);
                v8bf bl0 = *(const v8bf*)bl;
                v8bf bl1 = *(const v8bf*)(bl + 16);
                v16bf Bh = __builtin_shufflevector(bh0, bh1,
                    0,1,2,3,4,5,6,7,8,9,10,11,12,13,14,15);
                v16bf Bl = __builtin_shufflevector(bl0, bl1,
                    0,1,2,3,4,5,6,7,8,9,10,11,12,13,14,15);

                // bf16x3 split-product: near-fp32 accuracy
                acc = __builtin_amdgcn_wmma_f32_16x16x32_bf16(
                    false, Ah[kb], false, Bh, (short)0, acc, false, false);
                acc = __builtin_amdgcn_wmma_f32_16x16x32_bf16(
                    false, Al[kb], false, Bh, (short)0, acc, false, false);
                acc = __builtin_amdgcn_wmma_f32_16x16x32_bf16(
                    false, Ah[kb], false, Bl, (short)0, acc, false, false);
            }
            // fold this N-block into row partials: acc[r] = U[r + 8h][j*16 + c]
            #pragma unroll
            for (int r = 0; r < 8; ++r) {
                float t = trow[(r + 8 * h) * DIM + j * 16 + c];  // coalesced per half
                rp[r] += acc[r] * t;
            }
        }

        // reduce across the 16 lanes of each half (columns of the row)
        #pragma unroll
        for (int r = 0; r < 8; ++r) {
            float v = rp[r];
            v += __shfl_xor(v, 1);
            v += __shfl_xor(v, 2);
            v += __shfl_xor(v, 4);
            v += __shfl_xor(v, 8);
            if (c == 0) {   // lane 0 -> row r, lane 16 -> row r+8
                int row = tile * 16 + r + 8 * h;
                float resid = v - Wt[row] - Bs[row];
                lsum += resid * resid;
            }
        }
    }

    // ---- block-level deterministic reduction ----
    lsum += __shfl_xor(lsum, 1);
    lsum += __shfl_xor(lsum, 2);
    lsum += __shfl_xor(lsum, 4);
    lsum += __shfl_xor(lsum, 8);
    lsum += __shfl_xor(lsum, 16);
    if (lane == 0) wavesum[waveInBlk] = lsum;
    __syncthreads();
    if (threadIdx.x == 0) {
        float s = 0.0f;
        #pragma unroll
        for (int i = 0; i < 8; ++i) s += wavesum[i];
        partial[blockIdx.x] = s;
    }
}

__global__ __launch_bounds__(256) void finalize_kernel(
    const float* __restrict__ partial, int n, float* __restrict__ out, float scale)
{
    __shared__ float sh[256];
    float v = 0.0f;
    for (int i = threadIdx.x; i < n; i += blockDim.x) v += partial[i];
    sh[threadIdx.x] = v;
    __syncthreads();
    for (int s = 128; s > 0; s >>= 1) {
        if ((int)threadIdx.x < s) sh[threadIdx.x] += sh[threadIdx.x + s];
        __syncthreads();
    }
    if (threadIdx.x == 0) out[0] = sh[0] * scale;
}

extern "C" void kernel_launch(void* const* d_in, const int* in_sizes, int n_in,
                              void* d_out, int out_size, void* d_ws, size_t ws_size,
                              hipStream_t stream) {
    const float* S  = (const float*)d_in[0];   // source   (B, D)
    const float* T  = (const float*)d_in[1];   // target   (B, D)
    const float* Wt = (const float*)d_in[2];   // weight   (B,)
    const float* Bs = (const float*)d_in[3];   // bias     (B,)
    const float* Wm = (const float*)d_in[4];   // matrixOperator (D, D)
    float* out      = (float*)d_out;
    float* partial  = (float*)d_ws;

    const int Bn       = in_sizes[2];          // batch size from weight vector
    const int numTiles = Bn / 16;
    const int blocks   = 1024;

    bilinear_diag_kernel<<<blocks, 256, 0, stream>>>(S, T, Wt, Bs, Wm, partial, numTiles);
    finalize_kernel<<<1, 256, 0, stream>>>(partial, blocks, out, 1.0f / (float)Bn);
}
